// SkipConnection_57664230916961
// MI455X (gfx1250) — compile-verified, matrix operand order checked by reading, not run on hardware
//
#include <hip/hip_runtime.h>

// ---------------------------------------------------------------------------
// Feature detection (device pass only; host pass takes fallbacks)
// ---------------------------------------------------------------------------
#if defined(__AMDGCN__) && __has_builtin(__builtin_amdgcn_global_load_async_to_lds_b128) && \
    __has_builtin(__builtin_amdgcn_s_wait_asynccnt)
#define USE_ASYNC_LDS 1
#else
#define USE_ASYNC_LDS 0
#endif

#if defined(__AMDGCN__) && __has_builtin(__builtin_amdgcn_tensor_load_to_lds) && \
    __has_builtin(__builtin_amdgcn_s_wait_tensorcnt) && \
    !__has_include(<hip/amd_detail/amd_gfx1250_TDM.h>)
#define USE_TDM 1          // 5-arg ROCm 7.2 style builtin
#else
#define USE_TDM 0
#endif

#define AS1 __attribute__((address_space(1)))
#define AS3 __attribute__((address_space(3)))

// ---------------------------------------------------------------------------
// Types for CDNA5 WMMA bf16 / TDM descriptors / async b128 payload
// ---------------------------------------------------------------------------
typedef __bf16    v16bf  __attribute__((ext_vector_type(16)));
typedef float     v8f    __attribute__((ext_vector_type(8)));
typedef unsigned  v4u    __attribute__((ext_vector_type(4)));
typedef int       v8i    __attribute__((ext_vector_type(8)));
typedef int       v4i    __attribute__((ext_vector_type(4)));
typedef int       b128_t __attribute__((__vector_size__(16)));  // async-LDS payload type

#define B_  8
#define H_  8
#define N_  1024
#define HD_ 96
#define DIM_ 768
#define SCALE_ 0.10206207261596577f   // 96^-0.5
#define BN_EPS_ 1e-3f
#define APAD_ 8                        // 16B row pad -> conflict-free ds_load_b128

// ---------------------------------------------------------------------------
// bf16 <-> f32 helpers (round-to-nearest-even)
// ---------------------------------------------------------------------------
static __device__ __forceinline__ unsigned short f2bf(float f) {
  unsigned u = __float_as_uint(f);
  unsigned r = u + 0x7FFFu + ((u >> 16) & 1u);
  return (unsigned short)(r >> 16);
}
static __device__ __forceinline__ float bf2f(unsigned short b) {
  return __uint_as_float(((unsigned)b) << 16);
}

// ---------------------------------------------------------------------------
// 16x32 bf16 WMMA operand loaders (A, or B from n-major storage).
// ISA 16-bit layout: lane l: idx=l&15 is the row (A) / col (B),
// K = {h*8+0..7, 16+h*8+0..7}, h=l>>4 -> two contiguous 16B loads per lane.
// ---------------------------------------------------------------------------
static __device__ __forceinline__ v16bf load_op16(const unsigned short* __restrict__ base,
                                                  int ld, int row16, int k0) {
  int lane = threadIdx.x & 31;
  int idx  = lane & 15;
  int half = lane >> 4;
  const unsigned short* p = base + (size_t)(row16 + idx) * ld + k0 + half * 8;
  union { v16bf v; uint4 q[2]; } u;
  u.q[0] = *(const uint4*)(p);
  u.q[1] = *(const uint4*)(p + 16);
  return u.v;
}

// Same gather from an LDS-resident tile (emits ds_load_b128 x2).
static __device__ __forceinline__ v16bf load_op16_lds(const unsigned short* p0) {
  union { v16bf v; uint4 q[2]; } u;
  u.q[0] = *(const uint4*)(p0);
  u.q[1] = *(const uint4*)(p0 + 16);
  return u.v;
}

static __device__ __forceinline__ v8f wmma_bf16(v16bf a, v16bf b, v8f c) {
  return __builtin_amdgcn_wmma_f32_16x16x32_bf16(false, a, false, b, (short)0, c,
                                                 false, false);
}

// ---------------------------------------------------------------------------
// K0: transpose + convert f32 [rows][cols] -> bf16 [cols][rows]
// ---------------------------------------------------------------------------
__global__ void transpose_cvt_kernel(const float* __restrict__ in,
                                     unsigned short* __restrict__ out,
                                     int rows, int cols) {
  int i = blockIdx.x * 256 + threadIdx.x;
  if (i < rows * cols) {
    int r = i / cols, c = i - r * cols;
    out[(size_t)c * rows + r] = f2bf(in[(size_t)r * cols + c]);
  }
}

// ---------------------------------------------------------------------------
// K1: fused 3x3 SAME conv for q,k,v (per token 16x16x3), outputs:
//   qh,kh : bf16 [B,H,N,HD]   (A / B^T operand layout for QK^T)
//   vt    : bf16 [B,H,HD,N]   (n-major so attn@V B-operand is contiguous)
// ---------------------------------------------------------------------------
__global__ void conv_qkv_kernel(const float* __restrict__ q, const float* __restrict__ k,
                                const float* __restrict__ v, const float* __restrict__ wq,
                                const float* __restrict__ wk, const float* __restrict__ wv,
                                unsigned short* __restrict__ qh,
                                unsigned short* __restrict__ kh,
                                unsigned short* __restrict__ vt) {
  __shared__ float xin[3][DIM_];
  __shared__ float wgt[3][81];
  int token = blockIdx.x;
  int b = token >> 10, n = token & 1023;
  int t = threadIdx.x;
  size_t base = (size_t)token * DIM_;
  for (int i = t; i < DIM_; i += 256) {
    xin[0][i] = q[base + i];
    xin[1][i] = k[base + i];
    xin[2][i] = v[base + i];
  }
  if (t < 81) { wgt[0][t] = wq[t]; wgt[1][t] = wk[t]; wgt[2][t] = wv[t]; }
  __syncthreads();
#pragma unroll
  for (int j = 0; j < 3; ++j) {
    int f  = t + 256 * j;
    int r  = f / 48, cc = (f / 3) & 15, co = f % 3;
    int h  = f / HD_, hd = f % HD_;
    float acc0 = 0.f, acc1 = 0.f, acc2 = 0.f;
#pragma unroll
    for (int dr = -1; dr <= 1; ++dr) {
      int rr = r + dr;
      if (rr < 0 || rr > 15) continue;
#pragma unroll
      for (int dc = -1; dc <= 1; ++dc) {
        int c2 = cc + dc;
        if (c2 < 0 || c2 > 15) continue;
        int ib = (rr * 16 + c2) * 3;
        int wb = ((dr + 1) * 3 + (dc + 1)) * 9 + co;
#pragma unroll
        for (int ci = 0; ci < 3; ++ci) {
          acc0 += xin[0][ib + ci] * wgt[0][wb + ci * 3];
          acc1 += xin[1][ib + ci] * wgt[1][wb + ci * 3];
          acc2 += xin[2][ib + ci] * wgt[2][wb + ci * 3];
        }
      }
    }
    size_t ho = ((size_t)(b * H_ + h) * N_ + n) * HD_ + hd;
    qh[ho] = f2bf(acc0);
    kh[ho] = f2bf(acc1);
    vt[((size_t)(b * H_ + h) * HD_ + hd) * N_ + n] = f2bf(acc2);
  }
}

// ---------------------------------------------------------------------------
// K2: S[bh] = (Q[bh] @ K[bh]^T) * SCALE   -> bf16 [64][1024][1024]
// Wave: 16x64 tile, K = 96 (fully unrolled). 65536 waves -> 8192 blocks.
// ---------------------------------------------------------------------------
__global__ void __launch_bounds__(256) gemm_qk_kernel(
    const unsigned short* __restrict__ qh, const unsigned short* __restrict__ kh,
    unsigned short* __restrict__ S) {
  int wid = blockIdx.x * 8 + (threadIdx.x >> 5);
  int cg  = wid & 15;
  int rt  = (wid >> 4) & 63;
  int bh  = wid >> 10;
  const unsigned short* Qp = qh + (size_t)bh * N_ * HD_;
  const unsigned short* Kp = kh + (size_t)bh * N_ * HD_;
  unsigned short*       Op = S  + (size_t)bh * N_ * N_;
  int row = rt * 16, col = cg * 64;
  v8f acc0 = {}, acc1 = {}, acc2 = {}, acc3 = {};
#pragma unroll
  for (int k0 = 0; k0 < HD_; k0 += 32) {
    v16bf a = load_op16(Qp, HD_, row, k0);
    acc0 = wmma_bf16(a, load_op16(Kp, HD_, col + 0,  k0), acc0);
    acc1 = wmma_bf16(a, load_op16(Kp, HD_, col + 16, k0), acc1);
    acc2 = wmma_bf16(a, load_op16(Kp, HD_, col + 32, k0), acc2);
    acc3 = wmma_bf16(a, load_op16(Kp, HD_, col + 48, k0), acc3);
  }
  int lane = threadIdx.x & 31, idx = lane & 15, half = lane >> 4;
  v8f accs[4] = {acc0, acc1, acc2, acc3};
#pragma unroll
  for (int tt = 0; tt < 4; ++tt) {
#pragma unroll
    for (int r = 0; r < 8; ++r)
      Op[(size_t)(row + r + half * 8) * N_ + col + tt * 16 + idx] =
          f2bf(accs[tt][r] * SCALE_);
  }
}

// ---------------------------------------------------------------------------
// K3: in-place row softmax over 1024 elements (bf16). One block per row.
// ---------------------------------------------------------------------------
__global__ void softmax_rows_kernel(unsigned short* __restrict__ S) {
  __shared__ float redmax[8];
  __shared__ float redsum[8];
  unsigned short* p = S + (size_t)blockIdx.x * N_;
  int t = threadIdx.x;
  float v0 = bf2f(p[t]), v1 = bf2f(p[t + 256]),
        v2 = bf2f(p[t + 512]), v3 = bf2f(p[t + 768]);
  float m = fmaxf(fmaxf(v0, v1), fmaxf(v2, v3));
#pragma unroll
  for (int off = 16; off > 0; off >>= 1) m = fmaxf(m, __shfl_xor(m, off, 32));
  if ((t & 31) == 0) redmax[t >> 5] = m;
  __syncthreads();
  m = redmax[0];
#pragma unroll
  for (int w = 1; w < 8; ++w) m = fmaxf(m, redmax[w]);
  v0 = __expf(v0 - m); v1 = __expf(v1 - m);
  v2 = __expf(v2 - m); v3 = __expf(v3 - m);
  float s = v0 + v1 + v2 + v3;
#pragma unroll
  for (int off = 16; off > 0; off >>= 1) s += __shfl_xor(s, off, 32);
  if ((t & 31) == 0) redsum[t >> 5] = s;
  __syncthreads();
  s = 0.f;
#pragma unroll
  for (int w = 0; w < 8; ++w) s += redsum[w];
  float inv = 1.0f / s;
  p[t]       = f2bf(v0 * inv);
  p[t + 256] = f2bf(v1 * inv);
  p[t + 512] = f2bf(v2 * inv);
  p[t + 768] = f2bf(v3 * inv);
}

// ---------------------------------------------------------------------------
// K4: re-attention A2 = BN(softmax(S) @ W_re + b_re). Dominant GEMM, K=1024.
// Block = 8 waves sharing one 16x1024 A strip -> staged in LDS (async when
// available), rows padded 16B so ds_load_b128 column reads are conflict-free.
// B operands register-double-buffered to overlap loads with WMMA.
// ---------------------------------------------------------------------------
__global__ void __launch_bounds__(256) gemm_reatten_kernel(
    const unsigned short* __restrict__ S, const unsigned short* __restrict__ WreT,
    const float* __restrict__ b_re, const float* __restrict__ bn_mean,
    const float* __restrict__ bn_var, const float* __restrict__ bn_gamma,
    const float* __restrict__ bn_beta, unsigned short* __restrict__ A2) {
  __shared__ __align__(16) unsigned short Alds[16][N_ + APAD_];
  int wid0 = blockIdx.x * 8;
  int cg0  = wid0 & 15;
  int rt   = (wid0 >> 4) & 63;
  int bh   = wid0 >> 10;
  const unsigned short* Sp = S + (size_t)bh * N_ * N_ + (size_t)rt * 16 * N_;
  unsigned short*       Op = A2 + (size_t)bh * N_ * N_;

  // ---- stage A strip: 16 rows x 1024 cols = 2048 x 16B chunks, 8 per thread
  {
    int t = threadIdx.x;
#pragma unroll
    for (int i = 0; i < 8; ++i) {
      int c  = t + i * 256;
      int r  = c >> 7;            // 128 chunks per row
      int cc = c & 127;
      const unsigned short* gp = Sp + (size_t)r * N_ + cc * 8;
      unsigned short*       lp = &Alds[r][cc * 8];
#if USE_ASYNC_LDS
      __builtin_amdgcn_global_load_async_to_lds_b128(
          (AS1 b128_t*)(void*)gp, (AS3 b128_t*)(void*)lp, 0, 0);
#else
      *(uint4*)lp = *(const uint4*)gp;
#endif
    }
  }
#if USE_ASYNC_LDS
  __builtin_amdgcn_s_wait_asynccnt(0);
#endif
  __syncthreads();

  int wave = threadIdx.x >> 5;
  int col  = (cg0 + wave) * 64;
  int lane = threadIdx.x & 31, idx = lane & 15, half = lane >> 4;
  int row  = rt * 16;

  v8f acc0 = {}, acc1 = {}, acc2 = {}, acc3 = {};
  v16bf b0 = load_op16(WreT, N_, col + 0,  0);
  v16bf b1 = load_op16(WreT, N_, col + 16, 0);
  v16bf b2 = load_op16(WreT, N_, col + 32, 0);
  v16bf b3 = load_op16(WreT, N_, col + 48, 0);
  for (int k0 = 0; k0 < N_; k0 += 32) {
    int kn = (k0 + 32 < N_) ? k0 + 32 : k0;   // dummy-safe tail reload
    v16bf a  = load_op16_lds(&Alds[idx][k0 + half * 8]);
    v16bf n0 = load_op16(WreT, N_, col + 0,  kn);
    v16bf n1 = load_op16(WreT, N_, col + 16, kn);
    v16bf n2 = load_op16(WreT, N_, col + 32, kn);
    v16bf n3 = load_op16(WreT, N_, col + 48, kn);
    acc0 = wmma_bf16(a, b0, acc0);
    acc1 = wmma_bf16(a, b1, acc1);
    acc2 = wmma_bf16(a, b2, acc2);
    acc3 = wmma_bf16(a, b3, acc3);
    b0 = n0; b1 = n1; b2 = n2; b3 = n3;
  }

  v8f accs[4] = {acc0, acc1, acc2, acc3};
#pragma unroll
  for (int tt = 0; tt < 4; ++tt) {
    int p     = col + tt * 16 + idx;
    float rs  = rsqrtf(bn_var[p] + BN_EPS_) * bn_gamma[p];
    float add = b_re[p] - bn_mean[p];
    float bet = bn_beta[p];
#pragma unroll
    for (int r = 0; r < 8; ++r) {
      float val = (accs[tt][r] + add) * rs + bet;
      Op[(size_t)(row + r + half * 8) * N_ + p] = f2bf(val);
    }
  }
}

// ---------------------------------------------------------------------------
// K5: X = A2 @ V  -> [B,N,DIM] bf16 (head-interleaved). Wave: 16x96 tile.
// ---------------------------------------------------------------------------
__global__ void __launch_bounds__(256) gemm_av_kernel(
    const unsigned short* __restrict__ A2, const unsigned short* __restrict__ vt,
    unsigned short* __restrict__ Xf) {
  int wid = blockIdx.x * 8 + (threadIdx.x >> 5);
  int rt  = wid & 63;
  int bh  = wid >> 6;
  int b   = bh >> 3, h = bh & 7;
  const unsigned short* Ap = A2 + (size_t)bh * N_ * N_;
  const unsigned short* Vp = vt + (size_t)bh * HD_ * N_;
  int row = rt * 16;
  v8f acc[6] = {{}, {}, {}, {}, {}, {}};
  for (int k0 = 0; k0 < N_; k0 += 32) {
    v16bf a = load_op16(Ap, N_, row, k0);
#pragma unroll
    for (int tt = 0; tt < 6; ++tt)
      acc[tt] = wmma_bf16(a, load_op16(Vp, N_, tt * 16, k0), acc[tt]);
  }
  int lane = threadIdx.x & 31, idx = lane & 15, half = lane >> 4;
#pragma unroll
  for (int tt = 0; tt < 6; ++tt) {
    int hd = tt * 16 + idx;
#pragma unroll
    for (int r = 0; r < 8; ++r) {
      int n = row + r + half * 8;
      Xf[((size_t)(b * N_ + n)) * DIM_ + h * HD_ + hd] = f2bf(acc[tt][r]);
    }
  }
}

// ---------------------------------------------------------------------------
// K6: out = Xf @ w_proj + b_proj  (fp32 out). Block = 12 waves (384 thr) on
// one 16-row strip; A strip is contiguous 16*768 bf16 = 24KB -> one 1D TDM
// tensor_load_to_lds per block (when available), else cooperative copy.
// ---------------------------------------------------------------------------
__global__ void __launch_bounds__(384) gemm_proj_kernel(
    const unsigned short* __restrict__ Xf, const unsigned short* __restrict__ WpT,
    const float* __restrict__ b_proj, float* __restrict__ out) {
  __shared__ __align__(16) unsigned short Alds[16][DIM_];
  int rt  = blockIdx.x;
  int row = rt * 16;
  const unsigned short* Ap = Xf + (size_t)row * DIM_;   // contiguous 16x768

#if USE_TDM
  if (threadIdx.x < 32) {
    // Tensor DMA descriptor (D#), ISA ch.8: 1D tile of 12288 2-byte elements.
    unsigned lds_off = (unsigned)(unsigned long long)(AS3 unsigned short*)&Alds[0][0];
    unsigned long long ga = (unsigned long long)Ap;
    v4u g0 = { 1u,                                  // count=1, user mode
               lds_off,                              // lds_addr
               (unsigned)ga,                         // global_addr[31:0]
               (unsigned)((ga >> 32) & 0x01FFFFFFu) | (2u << 30) }; // addr[56:32]|type=2
    v8i g1 = { (int)(1u << 16),          // data_size=1 (2 bytes), no flags
               (int)(12288u << 16),      // tensor_dim0[15:0] in bits 63:48
               (int)(1u << 16),          // tensor_dim1 = 1
               (int)(12288u << 16),      // tile_dim0 = 12288 (bits 127:112)
               1,                        // tile_dim1 = 1
               12288,                    // tensor_dim0_stride[31:0]
               0, 0 };
    v4i g2 = { 0, 0, 0, 0 };
    v4i g3 = { 0, 0, 0, 0 };
    __builtin_amdgcn_tensor_load_to_lds(g0, g1, g2, g3, 0);
    __builtin_amdgcn_s_wait_tensorcnt(0);
  }
#else
  {
    int t = threadIdx.x;                 // 1536 x 16B chunks, 4 per thread
#pragma unroll
    for (int i = 0; i < 4; ++i) {
      int c  = t + i * 384;
      int r  = c / 96;                   // 96 chunks per row
      int cc = c - r * 96;
      *(uint4*)&Alds[r][cc * 8] = *(const uint4*)(Ap + (size_t)r * DIM_ + cc * 8);
    }
  }
#endif
  __syncthreads();

  int wave = threadIdx.x >> 5;           // 0..11
  int col  = wave * 64;
  int lane = threadIdx.x & 31, idx = lane & 15, half = lane >> 4;

  v8f acc0 = {}, acc1 = {}, acc2 = {}, acc3 = {};
  v16bf b0 = load_op16(WpT, DIM_, col + 0,  0);
  v16bf b1 = load_op16(WpT, DIM_, col + 16, 0);
  v16bf b2 = load_op16(WpT, DIM_, col + 32, 0);
  v16bf b3 = load_op16(WpT, DIM_, col + 48, 0);
  for (int k0 = 0; k0 < DIM_; k0 += 32) {
    int kn = (k0 + 32 < DIM_) ? k0 + 32 : k0;
    v16bf a  = load_op16_lds(&Alds[idx][k0 + half * 8]);
    v16bf n0 = load_op16(WpT, DIM_, col + 0,  kn);
    v16bf n1 = load_op16(WpT, DIM_, col + 16, kn);
    v16bf n2 = load_op16(WpT, DIM_, col + 32, kn);
    v16bf n3 = load_op16(WpT, DIM_, col + 48, kn);
    acc0 = wmma_bf16(a, b0, acc0);
    acc1 = wmma_bf16(a, b1, acc1);
    acc2 = wmma_bf16(a, b2, acc2);
    acc3 = wmma_bf16(a, b3, acc3);
    b0 = n0; b1 = n1; b2 = n2; b3 = n3;
  }

  v8f accs[4] = {acc0, acc1, acc2, acc3};
#pragma unroll
  for (int tt = 0; tt < 4; ++tt) {
    int p = col + tt * 16 + idx;
    float bias = b_proj[p];
#pragma unroll
    for (int r = 0; r < 8; ++r)
      out[(size_t)(row + r + half * 8) * DIM_ + p] = accs[tt][r] + bias;
  }
}

// ---------------------------------------------------------------------------
// Host-side launch
// ---------------------------------------------------------------------------
extern "C" void kernel_launch(void* const* d_in, const int* in_sizes, int n_in,
                              void* d_out, int out_size, void* d_ws, size_t ws_size,
                              hipStream_t stream) {
  const float* q        = (const float*)d_in[0];
  const float* k        = (const float*)d_in[1];
  const float* v        = (const float*)d_in[2];
  const float* wq       = (const float*)d_in[3];
  const float* wk       = (const float*)d_in[4];
  const float* wv       = (const float*)d_in[5];
  const float* w_re     = (const float*)d_in[6];
  const float* b_re     = (const float*)d_in[7];
  const float* bn_gamma = (const float*)d_in[8];
  const float* bn_beta  = (const float*)d_in[9];
  const float* bn_mean  = (const float*)d_in[10];
  const float* bn_var   = (const float*)d_in[11];
  const float* w_proj   = (const float*)d_in[12];
  const float* b_proj   = (const float*)d_in[13];
  float* out = (float*)d_out;

  char* ws = (char*)d_ws;
  size_t off = 0;
  auto alloc = [&](size_t bytes) -> void* {
    void* p = ws + off;
    off += (bytes + 255) & ~(size_t)255;
    return p;
  };
  const size_t BH = (size_t)B_ * H_;
  unsigned short* qh   = (unsigned short*)alloc(BH * N_ * HD_ * 2);          // 12 MB
  unsigned short* kh   = (unsigned short*)alloc(BH * N_ * HD_ * 2);          // 12 MB
  unsigned short* vt   = (unsigned short*)alloc(BH * HD_ * N_ * 2);          // 12 MB
  unsigned short* S    = (unsigned short*)alloc(BH * N_ * N_ * 2);           // 128 MB
  unsigned short* A2   = (unsigned short*)alloc(BH * N_ * N_ * 2);           // 128 MB
  unsigned short* Xf   = (unsigned short*)alloc((size_t)B_ * N_ * DIM_ * 2); // 12 MB
  unsigned short* WreT = (unsigned short*)alloc((size_t)N_ * N_ * 2);        // 2 MB
  unsigned short* WpT  = (unsigned short*)alloc((size_t)DIM_ * DIM_ * 2);    // ~1.1 MB

  transpose_cvt_kernel<<<(N_ * N_ + 255) / 256, 256, 0, stream>>>(w_re, WreT, N_, N_);
  transpose_cvt_kernel<<<(DIM_ * DIM_ + 255) / 256, 256, 0, stream>>>(w_proj, WpT, DIM_, DIM_);

  conv_qkv_kernel<<<B_ * N_, 256, 0, stream>>>(q, k, v, wq, wk, wv, qh, kh, vt);

  gemm_qk_kernel<<<8192, 256, 0, stream>>>(qh, kh, S);

  softmax_rows_kernel<<<(int)(BH * N_), 256, 0, stream>>>(S);

  gemm_reatten_kernel<<<8192, 256, 0, stream>>>(S, WreT, b_re, bn_mean, bn_var,
                                                bn_gamma, bn_beta, A2);

  gemm_av_kernel<<<512, 256, 0, stream>>>(A2, vt, Xf);

  gemm_proj_kernel<<<512, 384, 0, stream>>>(Xf, WpT, b_proj, out);
}
